// GAT_16844861735392
// MI455X (gfx1250) — matrix operand order repaired and checked by reading, not compiled, not run on hardware
//
#include <hip/hip_runtime.h>
#include <hip/hip_bf16.h>

typedef __attribute__((ext_vector_type(16))) _Float16 v16h;
typedef __attribute__((ext_vector_type(8)))  _Float16 v8h;
typedef __attribute__((ext_vector_type(8)))  float    v8f;

#define N_NODES 50000
#define N_EDGES 800000
#define E_TOT   (N_EDGES + N_NODES)   // with self loops
#define K_DIM   128

// ---------------------------------------------------------------------------
// WMMA GEMM: C[M,NCOLS] = A[M,128] x B[128,NCOLS]; fp32 in/out, f16 MACs.
// Block = 256 threads (8 waves), 64 rows per block, whole K staged in LDS.
// ---------------------------------------------------------------------------
template<int NCOLS>
__global__ __launch_bounds__(256)
void wmma_gemm128(const float* __restrict__ A, const float* __restrict__ B,
                  float* __restrict__ C, int M)
{
    constexpr int NT          = NCOLS / 16;   // 8 or 4 column tiles
    constexpr int CT_PER_WAVE = NT / 2;       // 4 or 2 tiles per wave
    __shared__ __align__(16) _Float16 Ah[64][136];     // 64 rows x K (+16B pad)
    __shared__ __align__(16) _Float16 Bt[NCOLS][128];  // weights, transposed

    const int tid     = threadIdx.x;
    const int wave    = tid >> 5;
    const int lane    = tid & 31;
    const int rowBase = blockIdx.x * 64;

    // stage B transposed: Bt[n][k] = B[k][n], fp32 -> f16
    for (int i = tid; i < K_DIM * NCOLS; i += 256) {
        int k = i / NCOLS, n = i % NCOLS;
        Bt[n][k] = (_Float16)B[i];
    }
    // stage A tile, fp32 -> f16 (zero-pad past M)
    for (int i = tid; i < 64 * (K_DIM / 4); i += 256) {
        int r  = i / (K_DIM / 4);
        int c4 = (i % (K_DIM / 4)) * 4;
        int gr = rowBase + r;
        float4 v = make_float4(0.f, 0.f, 0.f, 0.f);
        if (gr < M) v = *(const float4*)(A + (size_t)gr * K_DIM + c4);
        Ah[r][c4 + 0] = (_Float16)v.x;
        Ah[r][c4 + 1] = (_Float16)v.y;
        Ah[r][c4 + 2] = (_Float16)v.z;
        Ah[r][c4 + 3] = (_Float16)v.w;
    }
    __syncthreads();

    const int rt = wave >> 1;          // row tile 0..3
    const int ch = wave & 1;           // column half
    const int r0 = rt * 16;
    const int hs = lane >> 4;          // lane-half select
    const int ml = lane & 15;

    v8f acc[CT_PER_WAVE];
    #pragma unroll
    for (int t = 0; t < CT_PER_WAVE; ++t) acc[t] = (v8f){0,0,0,0,0,0,0,0};

    #pragma unroll
    for (int ks = 0; ks < 4; ++ks) {
        const int kb = ks * 32;
        // A fragment (16x32 f16): e<8 -> K=kb+hs*8+e ; e>=8 -> K=kb+16+hs*8+(e-8)
        v8h alo = *(const v8h*)&Ah[r0 + ml][kb + hs * 8];
        v8h ahi = *(const v8h*)&Ah[r0 + ml][kb + 16 + hs * 8];
        v16h afrag = __builtin_shufflevector(alo, ahi,
            0,1,2,3,4,5,6,7,8,9,10,11,12,13,14,15);
        #pragma unroll
        for (int t = 0; t < CT_PER_WAVE; ++t) {
            const int c0 = (ch * CT_PER_WAVE + t) * 16;
            // B fragment (32x16 f16): N=ml, e<8 -> K=kb+hs*8+e ; e>=8 -> K=kb+16+hs*8+(e-8)
            const _Float16* q = &Bt[c0 + ml][kb + hs * 8];
            v8h blo = *(const v8h*)q;
            v8h bhi = *(const v8h*)(q + 16);
            v16h bfrag = __builtin_shufflevector(blo, bhi,
                0,1,2,3,4,5,6,7,8,9,10,11,12,13,14,15);
            acc[t] = __builtin_amdgcn_wmma_f32_16x16x32_f16(
                false, afrag, false, bfrag, (short)0, acc[t], false, false);
        }
    }

    // C/D layout: VGPR v, lanes 0-15 -> M=v, lanes 16-31 -> M=v+8, N=lane%16
    #pragma unroll
    for (int t = 0; t < CT_PER_WAVE; ++t) {
        const int col = (ch * CT_PER_WAVE + t) * 16 + ml;
        #pragma unroll
        for (int v = 0; v < 8; ++v) {
            int row = rowBase + r0 + v + hs * 8;
            if (row < M) C[(size_t)row * NCOLS + col] = acc[t][v];
        }
    }
}

// ---------------------------------------------------------------------------
// helpers
// ---------------------------------------------------------------------------
__device__ __forceinline__ void atomicMaxFloat(float* addr, float val) {
    // monotonic int-key trick; valid with -inf initialization
    if (val >= 0.f) atomicMax((int*)addr, __float_as_int(val));
    else            atomicMin((unsigned int*)addr, __float_as_uint(val));
}

__device__ __forceinline__ float leaky_dot4(float4 l, float4 r, float4 a) {
    float acc = 0.f, v;
    v = l.x + r.x; v = v > 0.f ? v : 0.2f * v; acc += v * a.x;
    v = l.y + r.y; v = v > 0.f ? v : 0.2f * v; acc += v * a.y;
    v = l.z + r.z; v = v > 0.f ? v : 0.2f * v; acc += v * a.z;
    v = l.w + r.w; v = v > 0.f ? v : 0.2f * v; acc += v * a.w;
    return acc;
}

// ---------------------------------------------------------------------------
// init kernels
// ---------------------------------------------------------------------------
__global__ void init1(float* nmax, float* den, float* agg, float* red) {
    int t = blockIdx.x * blockDim.x + threadIdx.x;
    if (t < N_NODES * 8) { nmax[t] = -__builtin_inff(); den[t] = 0.f; }
    if (t < N_NODES * 128) agg[t] = 0.f;
    if (t < 4) red[t] = 0.f;
}

__global__ void init2(float* nmax, float* den, float* out) {
    int t = blockIdx.x * blockDim.x + threadIdx.x;
    if (t < N_NODES) { nmax[t] = -__builtin_inff(); den[t] = 0.f; }
    if (t < N_NODES * 64) out[t] = 0.f;
}

// ---------------------------------------------------------------------------
// layer 1 edge passes (8 heads x 16 ch)
// ---------------------------------------------------------------------------
__global__ __launch_bounds__(256)
void edge_logits1(const int* __restrict__ src, const int* __restrict__ dst,
                  const float* __restrict__ xl, const float* __restrict__ xr,
                  const float* __restrict__ att, float* __restrict__ logits,
                  float* __restrict__ nmax)
{
    long long t = (long long)blockIdx.x * 256 + threadIdx.x;
    if (t >= (long long)E_TOT * 8) return;
    int e = (int)(t >> 3), h = (int)(t & 7);
    int s = (e < N_EDGES) ? src[e] : (e - N_EDGES);
    int d = (e < N_EDGES) ? dst[e] : (e - N_EDGES);
    const float4* pl = (const float4*)(xl + (size_t)s * 128 + h * 16);
    const float4* pr = (const float4*)(xr + (size_t)d * 128 + h * 16);
    const float4* pa = (const float4*)(att + h * 16);
    float acc = 0.f;
    #pragma unroll
    for (int i = 0; i < 4; ++i) acc += leaky_dot4(pl[i], pr[i], pa[i]);
    logits[(size_t)e * 8 + h] = acc;
    atomicMaxFloat(&nmax[(size_t)d * 8 + h], acc);
}

__global__ __launch_bounds__(256)
void edge_agg1(const int* __restrict__ src, const int* __restrict__ dst,
               const float* __restrict__ xl, const float* __restrict__ logits,
               const float* __restrict__ nmax, float* __restrict__ den,
               float* __restrict__ agg)
{
    long long t = (long long)blockIdx.x * 256 + threadIdx.x;
    if (t >= (long long)E_TOT * 8) return;
    int e = (int)(t >> 3), h = (int)(t & 7);
    int s = (e < N_EDGES) ? src[e] : (e - N_EDGES);
    int d = (e < N_EDGES) ? dst[e] : (e - N_EDGES);
    float p = __expf(logits[(size_t)e * 8 + h] - nmax[(size_t)d * 8 + h]);
    atomicAdd(&den[(size_t)d * 8 + h], p);
    const float* xs = xl + (size_t)s * 128 + h * 16;
    float* ag = agg + (size_t)d * 128 + h * 16;
    #pragma unroll
    for (int c = 0; c < 16; ++c) atomicAdd(&ag[c], p * xs[c]);
}

// combine + global-LayerNorm partial reduction
__global__ __launch_bounds__(256)
void combine1(const float* __restrict__ agg, const float* __restrict__ den,
              const float* __restrict__ sk, const float* __restrict__ b1,
              const float* __restrict__ skb, float* __restrict__ hpre,
              float* __restrict__ red)
{
    int t = blockIdx.x * 256 + threadIdx.x;   // t < N*128 exactly
    int j = t & 127, n = t >> 7;
    float val = agg[t] / den[n * 8 + (j >> 4)] + b1[j] + sk[t] + skb[j];
    hpre[t] = val;
    __shared__ float ssum[256], ssq[256];
    ssum[threadIdx.x] = val; ssq[threadIdx.x] = val * val;
    __syncthreads();
    for (int o = 128; o > 0; o >>= 1) {
        if (threadIdx.x < o) {
            ssum[threadIdx.x] += ssum[threadIdx.x + o];
            ssq[threadIdx.x]  += ssq[threadIdx.x + o];
        }
        __syncthreads();
    }
    if (threadIdx.x == 0) { atomicAdd(&red[0], ssum[0]); atomicAdd(&red[1], ssq[0]); }
}

__global__ __launch_bounds__(256)
void norm_elu(const float* __restrict__ hpre, const float* __restrict__ red,
              const float* __restrict__ lnw, const float* __restrict__ lnb,
              float* __restrict__ h)
{
    int t = blockIdx.x * 256 + threadIdx.x;
    const float inv_cnt = 1.0f / (float)(N_NODES * 128);
    float mu  = red[0] * inv_cnt;
    float var = red[1] * inv_cnt - mu * mu;
    float is  = rsqrtf(var + 1e-5f);
    int j = t & 127;
    float y = lnw[j] * (hpre[t] - mu) * is + lnb[j];
    h[t] = y > 0.f ? y : (__expf(y) - 1.f);   // ELU
}

// ---------------------------------------------------------------------------
// layer 2 edge passes (1 head x 64 ch)
// ---------------------------------------------------------------------------
__global__ __launch_bounds__(256)
void edge_logits2(const int* __restrict__ src, const int* __restrict__ dst,
                  const float* __restrict__ xl, const float* __restrict__ xr,
                  const float* __restrict__ att, float* __restrict__ logits,
                  float* __restrict__ nmax)
{
    int e = blockIdx.x * 256 + threadIdx.x;
    if (e >= E_TOT) return;
    int s = (e < N_EDGES) ? src[e] : (e - N_EDGES);
    int d = (e < N_EDGES) ? dst[e] : (e - N_EDGES);
    const float4* pl = (const float4*)(xl + (size_t)s * 64);
    const float4* pr = (const float4*)(xr + (size_t)d * 64);
    const float4* pa = (const float4*)att;
    float acc = 0.f;
    #pragma unroll
    for (int i = 0; i < 16; ++i) acc += leaky_dot4(pl[i], pr[i], pa[i]);
    logits[e] = acc;
    atomicMaxFloat(&nmax[d], acc);
}

__global__ __launch_bounds__(256)
void edge_agg2(const int* __restrict__ src, const int* __restrict__ dst,
               const float* __restrict__ xl, const float* __restrict__ logits,
               const float* __restrict__ nmax, float* __restrict__ den,
               float* __restrict__ out)
{
    long long t = (long long)blockIdx.x * 256 + threadIdx.x;
    if (t >= (long long)E_TOT * 4) return;
    int e = (int)(t >> 2), g = (int)(t & 3);
    int s = (e < N_EDGES) ? src[e] : (e - N_EDGES);
    int d = (e < N_EDGES) ? dst[e] : (e - N_EDGES);
    float p = __expf(logits[e] - nmax[d]);
    if (g == 0) atomicAdd(&den[d], p);
    const float* xs = xl + (size_t)s * 64 + g * 16;
    float* ag = out + (size_t)d * 64 + g * 16;
    #pragma unroll
    for (int c = 0; c < 16; ++c) atomicAdd(&ag[c], p * xs[c]);
}

__global__ __launch_bounds__(256)
void finalize2(float* __restrict__ out, const float* __restrict__ den,
               const float* __restrict__ sk, const float* __restrict__ b2,
               const float* __restrict__ skb)
{
    int t = blockIdx.x * 256 + threadIdx.x;   // t < N*64 exactly
    int n = t >> 6, c = t & 63;
    out[t] = out[t] / den[n] + b2[c] + sk[t] + skb[c];
}

// ---------------------------------------------------------------------------
// launch
// ---------------------------------------------------------------------------
extern "C" void kernel_launch(void* const* d_in, const int* in_sizes, int n_in,
                              void* d_out, int out_size, void* d_ws, size_t ws_size,
                              hipStream_t stream)
{
    const float* x    = (const float*)d_in[0];
    const int*   ei   = (const int*)d_in[1];
    const int*   src  = ei;
    const int*   dst  = ei + N_EDGES;
    const float* W1l  = (const float*)d_in[2];
    const float* W1r  = (const float*)d_in[3];
    const float* att1 = (const float*)d_in[4];
    const float* b1   = (const float*)d_in[5];
    const float* s1W  = (const float*)d_in[6];
    const float* s1b  = (const float*)d_in[7];
    const float* lnw  = (const float*)d_in[8];
    const float* lnb  = (const float*)d_in[9];
    const float* W2l  = (const float*)d_in[10];
    const float* W2r  = (const float*)d_in[11];
    const float* att2 = (const float*)d_in[12];
    const float* b2   = (const float*)d_in[13];
    const float* s2W  = (const float*)d_in[14];
    const float* s2b  = (const float*)d_in[15];
    float* out = (float*)d_out;

    // workspace layout (floats), buffers reused across phases
    const size_t N = N_NODES;
    float* ws   = (float*)d_ws;
    float* XL1  = ws;                         // N*128   (later: H)
    float* XR1  = XL1 + N * 128;              // N*128   (later: HPRE)
    float* SK1  = XR1 + N * 128;              // N*128   (later: SK2)
    float* AGG1 = SK1 + N * 128;              // N*128   (later: XL2 | XR2)
    float* LOG1 = AGG1 + N * 128;             // E_TOT*8 (later: LOG2)
    float* MAX1 = LOG1 + (size_t)E_TOT * 8;   // N*8     (later: MAX2)
    float* DEN1 = MAX1 + N * 8;               // N*8     (later: DEN2)
    float* RED  = DEN1 + N * 8;               // 4

    dim3 b(256);
    const int gM = (N_NODES + 63) / 64;

    // ---- layer 1: dense transforms (WMMA) ----
    wmma_gemm128<128><<<gM, b, 0, stream>>>(x, W1l, XL1, N_NODES);
    wmma_gemm128<128><<<gM, b, 0, stream>>>(x, W1r, XR1, N_NODES);
    wmma_gemm128<128><<<gM, b, 0, stream>>>(x, s1W, SK1, N_NODES);

    // ---- layer 1: segment softmax + aggregation ----
    init1<<<(N_NODES * 128 + 255) / 256, b, 0, stream>>>(MAX1, DEN1, AGG1, RED);
    long long tA = (long long)E_TOT * 8;
    edge_logits1<<<(unsigned)((tA + 255) / 256), b, 0, stream>>>(src, dst, XL1, XR1, att1, LOG1, MAX1);
    edge_agg1<<<(unsigned)((tA + 255) / 256), b, 0, stream>>>(src, dst, XL1, LOG1, MAX1, DEN1, AGG1);

    // ---- combine + graph-LayerNorm + ELU ----
    float* HPRE = XR1;
    float* H    = XL1;
    combine1<<<(N_NODES * 128) / 256, b, 0, stream>>>(AGG1, DEN1, SK1, b1, s1b, HPRE, RED);
    norm_elu<<<(N_NODES * 128) / 256, b, 0, stream>>>(HPRE, RED, lnw, lnb, H);

    // ---- layer 2: dense transforms (WMMA) ----
    float* XL2 = AGG1;
    float* XR2 = AGG1 + N * 64;
    float* SK2 = SK1;
    wmma_gemm128<64><<<gM, b, 0, stream>>>(H, W2l, XL2, N_NODES);
    wmma_gemm128<64><<<gM, b, 0, stream>>>(H, W2r, XR2, N_NODES);
    wmma_gemm128<64><<<gM, b, 0, stream>>>(H, s2W, SK2, N_NODES);

    // ---- layer 2: segment softmax + aggregation ----
    float* LOG2 = LOG1; float* MAX2 = MAX1; float* DEN2 = DEN1;
    init2<<<(N_NODES * 64 + 255) / 256, b, 0, stream>>>(MAX2, DEN2, out);
    edge_logits2<<<(E_TOT + 255) / 256, b, 0, stream>>>(src, dst, XL2, XR2, att2, LOG2, MAX2);
    long long tB = (long long)E_TOT * 4;
    edge_agg2<<<(unsigned)((tB + 255) / 256), b, 0, stream>>>(src, dst, XL2, LOG2, MAX2, DEN2, out);
    finalize2<<<(N_NODES * 64) / 256, b, 0, stream>>>(out, DEN2, SK2, b2, s2b);
}
